// EvoformerIteration_9208409882906
// MI455X (gfx1250) — matrix operand order repaired and checked by reading, not compiled, not run on hardware
//
#include <hip/hip_runtime.h>
#include <hip/hip_bf16.h>

typedef __bf16 bf16;
typedef __attribute__((ext_vector_type(16))) __bf16 bf16x16;
typedef __attribute__((ext_vector_type(8)))  __bf16 bf16x8;
typedef __attribute__((ext_vector_type(8)))  float  f32x8;

__device__ __forceinline__ f32x8 wmma_bf16f32(bf16x16 a, bf16x16 b, f32x8 c) {
  return __builtin_amdgcn_wmma_f32_16x16x32_bf16(false, a, false, b, (short)0, c, false, false);
}
__device__ __forceinline__ bf16x16 join16(bf16x8 lo, bf16x8 hi) {
  bf16x16 r;
#pragma unroll
  for (int i = 0; i < 8; i++) { r[i] = lo[i]; r[8 + i] = hi[i]; }
  return r;
}

// ---------------- small helper kernels ----------------

__global__ void k_cvt(const float* __restrict__ s, bf16* __restrict__ d, int n) {
  int i = blockIdx.x * 256 + threadIdx.x;
  if (i < n) d[i] = (bf16)s[i];
}

// LayerNorm over last dim C, output bf16. perm=1: source row = swapped (256x256) row id.
__global__ void k_ln(const float* __restrict__ x, const float* __restrict__ g,
                     const float* __restrict__ o, bf16* __restrict__ y, int C, int perm) {
  int m = blockIdx.x;
  int src = perm ? (((m & 255) << 8) | (m >> 8)) : m;
  const float* row = x + (size_t)src * C;
  float s = 0.f, s2 = 0.f;
  for (int c = threadIdx.x; c < C; c += 128) { float v = row[c]; s += v; s2 += v * v; }
  __shared__ float sh0[128], sh1[128];
  sh0[threadIdx.x] = s; sh1[threadIdx.x] = s2; __syncthreads();
  for (int st = 64; st > 0; st >>= 1) {
    if ((int)threadIdx.x < st) { sh0[threadIdx.x] += sh0[threadIdx.x + st]; sh1[threadIdx.x] += sh1[threadIdx.x + st]; }
    __syncthreads();
  }
  float mean = sh0[0] / C;
  float var = sh1[0] / C - mean * mean;
  float rs = rsqrtf(var + 1e-5f);
  for (int c = threadIdx.x; c < C; c += 128) {
    float v = (row[c] - mean) * rs * g[c] + o[c];
    y[(size_t)m * C + c] = (bf16)v;
  }
}

// channel-major permute: out[c][a][b] = in[(a*256+b)*C + c]  (tsrc=1: in[(b*256+a)*C + c])
__global__ void k_permcm(const float* __restrict__ in, bf16* __restrict__ out, int C, int tsrc) {
  long long o = (long long)blockIdx.x * 256 + threadIdx.x;
  long long total = (long long)65536 * C;
  if (o >= total) return;
  int c = (int)(o >> 16);
  int r = (int)(o & 65535);
  int a = r >> 8, bb = r & 255;
  long long src = tsrc ? ((long long)(bb * 256 + a) * C + c) : ((long long)(a * 256 + bb) * C + c);
  out[o] = (bf16)in[src];
}

// OPM operand transpose: out[(i*32+c)*256 + s] = in[(s*256+i)*32 + c]
__global__ void k_permopm(const float* __restrict__ in, bf16* __restrict__ out) {
  int o = blockIdx.x * 256 + threadIdx.x;
  if (o >= 8192 * 256) return;
  int sdim = o & 255, row = o >> 8;
  int i = row >> 5, c = row & 31;
  out[o] = (bf16)in[((long long)(sdim * 256 + i)) * 32 + c];
}

// r[i*256+j] = 1/(1e-3 + sum_s mask[s,i]*mask[s,j])
__global__ void k_rnorm(const float* __restrict__ mask, float* __restrict__ r) {
  int ij = blockIdx.x * 256 + threadIdx.x;
  int i = ij >> 8, j = ij & 255;
  float s = 0.f;
  for (int ss = 0; ss < 256; ss++) s += mask[ss * 256 + i] * mask[ss * 256 + j];
  r[ij] = 1.f / (1e-3f + s);
}

// ---------------- generic bf16 WMMA GEMM ----------------

enum { GM_F32 = 0, GM_ADDF32 = 1, GM_BF16 = 2, GM_ADD_SWAP = 3, GM_OPM = 4 };
enum { ACT_NONE = 0, ACT_RELU = 1, ACT_SIG = 2 };

struct GemmArgs {
  const bf16* A; const bf16* B;
  const float* bias; const float* rowscale; const float* mul;
  float* outf; bf16* outb;
  long long a_z, b_z, out_z, ldm;
  int ldn, M, N, K;
  float alpha;
  int b_nmajor, mode, act;
};

__global__ void __launch_bounds__(128) k_gemm(GemmArgs g) {
  __shared__ __align__(16) bf16 Bt[64][40];
  const int z = blockIdx.z;
  const bf16* A = g.A + (long long)z * g.a_z;
  const bf16* B = g.B + (long long)z * g.b_z;
  const long long oz = (long long)z * g.out_z;
  const int n0 = blockIdx.x * 64;
  const int m0 = blockIdx.y * 64;
  const int wave = threadIdx.x >> 5;
  const int lane = threadIdx.x & 31;
  const int lm = lane & 15, kh = lane >> 4;
  const int fullN = (n0 + 64 <= g.N);    // block-uniform: vectorized staging path
  f32x8 acc[4] = {};
  for (int kc = 0; kc < g.K; kc += 32) {
    // ---- stage B tile into LDS, N-major: Bt[n_local][k_local] ----
    if (fullN) {
      if (g.b_nmajor) {
#pragma unroll
        for (int it = 0; it < 2; it++) {
          int idx = threadIdx.x + it * 128;          // 256 chunks of 8 along K
          int nl = idx >> 2, kg = idx & 3;
          const bf16* src = B + (long long)(n0 + nl) * g.K + kc + kg * 8;
          *(bf16x8*)&Bt[nl][kg * 8] = *(const bf16x8*)src;
        }
      } else {
#pragma unroll
        for (int it = 0; it < 2; it++) {
          int idx = threadIdx.x + it * 128;          // 256 chunks of 8 along N
          int kk = idx >> 3, ng = idx & 7;
          bf16x8 v = *(const bf16x8*)(B + (long long)(kc + kk) * g.N + n0 + ng * 8);
#pragma unroll
          for (int j = 0; j < 8; j++) Bt[ng * 8 + j][kk] = v[j];
        }
      }
    } else {
      for (int idx = threadIdx.x; idx < 64 * 32; idx += 128) {
        int nl = idx >> 5, kk = idx & 31;
        int n = n0 + nl;
        bf16 v = (bf16)0.f;
        if (n < g.N)
          v = g.b_nmajor ? B[(long long)n * g.K + kc + kk]
                         : B[(long long)(kc + kk) * g.N + n];
        Bt[nl][kk] = v;
      }
    }
    __syncthreads();
    bf16x16 a;
    {
      int r = m0 + wave * 16 + lm;
      if (r < g.M) {
        const bf16* ap = A + (long long)r * g.K + kc + kh * 8;
        a = join16(*(const bf16x8*)ap, *(const bf16x8*)(ap + 16));
        if (kc + 32 < g.K) __builtin_prefetch(ap + 32, 0, 1);   // global_prefetch next A chunk
      } else {
#pragma unroll
        for (int i = 0; i < 16; i++) a[i] = (bf16)0.f;
      }
    }
    if (fullN && (kc + 32 < g.K)) {
      // prefetch next B chunk
      const bf16* pf = g.b_nmajor
          ? (B + (long long)(n0 + (threadIdx.x >> 1)) * g.K + kc + 32)
          : (B + (long long)(kc + 32 + (threadIdx.x & 31)) * g.N + n0);
      __builtin_prefetch(pf, 0, 1);
    }
#pragma unroll
    for (int nt = 0; nt < 4; nt++) {
      const bf16* bp = &Bt[nt * 16 + lm][kh * 8];
      bf16x16 b = join16(*(const bf16x8*)bp, *(const bf16x8*)(bp + 16));
      acc[nt] = wmma_bf16f32(a, b, acc[nt]);
    }
    __syncthreads();
  }
  // epilogue
#pragma unroll
  for (int nt = 0; nt < 4; nt++) {
#pragma unroll
    for (int v = 0; v < 8; v++) {
      int row = m0 + wave * 16 + v + 8 * kh;
      int n = n0 + nt * 16 + lm;
      if (row < g.M && n < g.N) {
        float val = g.alpha * acc[nt][v] + (g.bias ? g.bias[n] : 0.f);
        if (g.act == ACT_RELU) val = fmaxf(val, 0.f);
        else if (g.act == ACT_SIG) val = 1.f / (1.f + __expf(-val));
        if (g.rowscale) val *= g.rowscale[row];
        if (g.mul) val *= g.mul[(long long)row * g.N + n];
        if (g.mode == GM_F32)
          g.outf[oz + (long long)row * g.ldm + (long long)n * g.ldn] = val;
        else if (g.mode == GM_ADDF32)
          g.outf[oz + (long long)row * g.ldm + (long long)n * g.ldn] += val;
        else if (g.mode == GM_BF16)
          g.outb[oz + (long long)row * g.ldm + (long long)n * g.ldn] = (bf16)val;
        else if (g.mode == GM_ADD_SWAP)
          g.outf[(long long)((row & 255) * 256 + (row >> 8)) * g.ldm + n] += val;
        else { // GM_OPM: m=(i,c) n=(j,e) -> [i,j, c*32+e]
          int i = row >> 5, c = row & 31, j = n >> 5, e = n & 31;
          g.outb[((long long)(i * 256 + j)) * 1024 + c * 32 + e] = (bf16)val;
        }
      }
    }
  }
}

// ---------------- fused gated attention (flash-style, K=Q=256, D=32) ----------------

__global__ void __launch_bounds__(128) k_flash(
    const bf16* __restrict__ Q, const bf16* __restrict__ K, const bf16* __restrict__ V,
    const float* __restrict__ mask, int maskT,
    const float* __restrict__ nb, int nH,
    const float* __restrict__ gate, bf16* __restrict__ out, int cols) {
  const int b = blockIdx.z, h = blockIdx.y;
  const int wave = threadIdx.x >> 5, lane = threadIdx.x & 31;
  const int lm = lane & 15, kh = lane >> 4;
  const int qw0 = blockIdx.x * 64 + wave * 16;
  __shared__ __align__(16) bf16 Vt[32][264];
  __shared__ __align__(16) bf16 P[4][16][264];
  // stage V transposed: Vt[d][k]  (vectorized global side)
#pragma unroll
  for (int it = 0; it < 8; it++) {
    int idx = threadIdx.x + it * 128;                // 1024 chunks of 8 along d
    int kk = idx >> 2, dg = idx & 3;
    bf16x8 v = *(const bf16x8*)(V + ((long long)b * 256 + kk) * cols + h * 32 + dg * 8);
#pragma unroll
    for (int j = 0; j < 8; j++) Vt[dg * 8 + j][kk] = v[j];
  }
  __syncthreads();
  // Q fragment (A layout, 16x32)
  bf16x16 aq;
  {
    const bf16* qp = Q + ((long long)b * 256 + qw0 + lm) * cols + h * 32 + kh * 8;
    aq = join16(*(const bf16x8*)qp, *(const bf16x8*)(qp + 16));
  }
  // S = Q K^T  (16 x 256)
  f32x8 s[16];
#pragma unroll
  for (int t = 0; t < 16; t++) {
    const bf16* kp = K + ((long long)b * 256 + t * 16 + lm) * cols + h * 32 + kh * 8;
    bf16x16 bk = join16(*(const bf16x8*)kp, *(const bf16x8*)(kp + 16));
    f32x8 zz = {};
    s[t] = wmma_bf16f32(aq, bk, zz);
  }
  // bias + nonbatched bias + clip
#pragma unroll
  for (int t = 0; t < 16; t++) {
    int k = t * 16 + lm;
    float bv = 1e9f * ((maskT ? mask[(long long)k * 256 + b] : mask[(long long)b * 256 + k]) - 1.f);
#pragma unroll
    for (int v = 0; v < 8; v++) {
      float val = s[t][v] + bv;
      if (nb) val += nb[((long long)(qw0 + v + 8 * kh) * 256 + k) * nH + h];
      s[t][v] = fminf(fmaxf(val, -1e8f), 1e8f);
    }
  }
  // softmax over rows (row spread across the 16 lanes of a half-wave)
#pragma unroll
  for (int v = 0; v < 8; v++) {
    float mx = -3e38f;
#pragma unroll
    for (int t = 0; t < 16; t++) mx = fmaxf(mx, s[t][v]);
    mx = fmaxf(mx, __shfl_xor(mx, 1, 32));
    mx = fmaxf(mx, __shfl_xor(mx, 2, 32));
    mx = fmaxf(mx, __shfl_xor(mx, 4, 32));
    mx = fmaxf(mx, __shfl_xor(mx, 8, 32));
    float sum = 0.f;
#pragma unroll
    for (int t = 0; t < 16; t++) { float p = __expf(s[t][v] - mx); s[t][v] = p; sum += p; }
    sum += __shfl_xor(sum, 1, 32);
    sum += __shfl_xor(sum, 2, 32);
    sum += __shfl_xor(sum, 4, 32);
    sum += __shfl_xor(sum, 8, 32);
    float inv = 1.f / sum;
    int m = v + 8 * kh;
#pragma unroll
    for (int t = 0; t < 16; t++) P[wave][m][t * 16 + lm] = (bf16)(s[t][v] * inv);
  }
  __syncthreads();
  // O = P V  (16 x 32)
  f32x8 oacc[2] = {};
#pragma unroll
  for (int kc = 0; kc < 8; kc++) {
    const bf16* pp = &P[wave][lm][kc * 32 + kh * 8];
    bf16x16 ap = join16(*(const bf16x8*)pp, *(const bf16x8*)(pp + 16));
#pragma unroll
    for (int t2 = 0; t2 < 2; t2++) {
      const bf16* vp = &Vt[t2 * 16 + lm][kc * 32 + kh * 8];
      bf16x16 bv = join16(*(const bf16x8*)vp, *(const bf16x8*)(vp + 16));
      oacc[t2] = wmma_bf16f32(ap, bv, oacc[t2]);
    }
  }
  // gate + store bf16
#pragma unroll
  for (int t2 = 0; t2 < 2; t2++) {
#pragma unroll
    for (int v = 0; v < 8; v++) {
      int q = qw0 + v + 8 * kh;
      int d = t2 * 16 + lm;
      long long oi = ((long long)b * 256 + q) * cols + h * 32 + d;
      out[oi] = (bf16)(oacc[t2][v] * gate[oi]);
    }
  }
}

// ---------------- host orchestration ----------------

extern "C" void kernel_launch(void* const* d_in, const int* in_sizes, int n_in,
                              void* d_out, int out_size, void* d_ws, size_t ws_size,
                              hipStream_t stream) {
  (void)in_sizes; (void)n_in; (void)out_size; (void)ws_size;
  const float* msa_in = (const float*)d_in[0];
  const float* pair_in = (const float*)d_in[1];
  const float* msa_mask = (const float*)d_in[2];
  const float* pair_mask = (const float*)d_in[3];
  auto F = [&](int i) -> const float* { return (const float*)d_in[i]; };

  float* msa = (float*)d_out;                 // [65536,256]
  float* pair = (float*)d_out + 16777216LL;   // [65536,128]
  hipMemcpyAsync(msa, msa_in, 16777216ull * 4, hipMemcpyDeviceToDevice, stream);
  hipMemcpyAsync(pair, pair_in, 8388608ull * 4, hipMemcpyDeviceToDevice, stream);

  char* ws = (char*)d_ws;
  const size_t MB = 1ull << 20;
  bf16* WB = (bf16*)(ws);                     // weights bf16 (8MB)
  bf16* LNB = (bf16*)(ws + 8 * MB);           // LN output bf16 (34MB)
  bf16* PZ = (bf16*)(ws + 42 * MB);           // LN(pair) for nb in stage mr (17MB)
  char* pQB = ws + 59 * MB;                   // 34MB
  char* pKB = ws + 93 * MB;                   // 34MB
  char* pVB = ws + 127 * MB;                  // 34MB
  char* pGATE = ws + 161 * MB;                // 68MB
  char* pWG = ws + 229 * MB;                  // 34MB
  float* NBf = (float*)(ws + 263 * MB);       // 3MB
  float* RNf = (float*)(ws + 266 * MB);       // 1MB
  bf16* QB = (bf16*)pQB; bf16* KB = (bf16*)pKB; bf16* VB = (bf16*)pVB;
  float* GATEf = (float*)pGATE;
  bf16* WGb = (bf16*)pWG;
  // stage-safe aliases:
  bf16* BIGH = (bf16*)pQB;                    // 65536x1024 bf16 (transition hidden / OPM T)
  float* opmL = (float*)pGATE;                // 65536x32 f32
  float* opmR = (float*)(pGATE + 17 * MB);
  bf16* opmLT = (bf16*)pWG;                   // 8192x256 bf16
  bf16* opmRT = (bf16*)(pWG + 9 * MB);
  float* triTmp = (float*)pQB;                // 65536x128 f32
  bf16* triA = (bf16*)pKB;                    // 128 x 256x256 bf16
  bf16* triB = (bf16*)pVB;
  float* triO = (float*)pWG;                  // 65536x128 f32

  size_t wcur = 0;
  auto cw = [&](int idx, size_t n) -> const bf16* {
    bf16* dst = WB + wcur; wcur += (n + 127) & ~(size_t)127;
    k_cvt<<<dim3((unsigned)((n + 255) / 256)), 256, 0, stream>>>(F(idx), dst, (int)n);
    return dst;
  };
  // weight conversions (indices per setup_inputs flattening order)
  const bf16 *mr_qw = cw(4, 65536), *mr_kw = cw(5, 65536), *mr_vw = cw(6, 65536),
             *mr_gw = cw(7, 65536), *mr_ow = cw(9, 65536), *mr_f2d = cw(15, 1024);
  const bf16 *mc_qw = cw(16, 65536), *mc_kw = cw(17, 65536), *mc_vw = cw(18, 65536),
             *mc_gw = cw(19, 65536), *mc_ow = cw(21, 65536);
  const bf16 *mt_w1 = cw(27, 262144), *mt_w2 = cw(29, 262144);
  const bf16 *opm_lw = cw(33, 8192), *opm_rw = cw(35, 8192), *opm_ow = cw(37, 131072);
  const bf16 *tmo_lw = cw(41, 16384), *tmo_rw = cw(43, 16384), *tmo_lgw = cw(45, 16384),
             *tmo_rgw = cw(47, 16384), *tmo_ow = cw(51, 16384), *tmo_gw = cw(53, 16384);
  const bf16 *tmi_lw = cw(57, 16384), *tmi_rw = cw(59, 16384), *tmi_lgw = cw(61, 16384),
             *tmi_rgw = cw(63, 16384), *tmi_ow = cw(67, 16384), *tmi_gw = cw(69, 16384);
  const bf16 *tas_qw = cw(71, 16384), *tas_kw = cw(72, 16384), *tas_vw = cw(73, 16384),
             *tas_gw = cw(74, 16384), *tas_ow = cw(76, 16384), *tas_f2d = cw(80, 512);
  const bf16 *tae_qw = cw(81, 16384), *tae_kw = cw(82, 16384), *tae_vw = cw(83, 16384),
             *tae_gw = cw(84, 16384), *tae_ow = cw(86, 16384), *tae_f2d = cw(90, 512);
  const bf16 *pt_w1 = cw(93, 65536), *pt_w2 = cw(95, 65536);

  auto gemm = [&](const bf16* A, const bf16* B, const float* bias,
                  int M, int N, int K, float alpha, int act, int mode,
                  float* outf, bf16* outb, long long ldm, int ldn,
                  const float* rowscale, const float* mul, int b_nmajor,
                  int batches, long long az, long long bz, long long ozz) {
    GemmArgs ga;
    ga.A = A; ga.B = B; ga.bias = bias; ga.rowscale = rowscale; ga.mul = mul;
    ga.outf = outf; ga.outb = outb;
    ga.a_z = az; ga.b_z = bz; ga.out_z = ozz; ga.ldm = ldm; ga.ldn = ldn;
    ga.M = M; ga.N = N; ga.K = K; ga.alpha = alpha;
    ga.b_nmajor = b_nmajor; ga.mode = mode; ga.act = act;
    dim3 grid((N + 63) / 64, (M + 63) / 64, batches);
    k_gemm<<<grid, 128, 0, stream>>>(ga);
  };
  auto ln = [&](const float* x, int gi, int oi, bf16* y, int M, int C, int perm) {
    k_ln<<<dim3((unsigned)M), 128, 0, stream>>>(x, F(gi), F(oi), y, C, perm);
  };
  auto flash = [&](const bf16* Qp, const bf16* Kp, const bf16* Vp,
                   const float* maskp, int maskT, const float* nbp, int nH,
                   const float* gatep, bf16* outp, int H, int cols) {
    k_flash<<<dim3(4, H, 256), 128, 0, stream>>>(Qp, Kp, Vp, maskp, maskT, nbp, nH, gatep, outp, cols);
  };
  const float QS = 0.17677669529663687f; // 1/sqrt(32)

  // ---- 1. MSA row attention with pair bias ----
  ln(msa, 11, 12, LNB, 65536, 256, 0);
  ln(pair, 13, 14, PZ, 65536, 128, 0);
  gemm(PZ, mr_f2d, nullptr, 65536, 8, 128, 1.f, ACT_NONE, GM_F32, NBf, nullptr, 8, 1, nullptr, nullptr, 0, 1, 0, 0, 0);
  gemm(LNB, mr_qw, nullptr, 65536, 256, 256, QS, ACT_NONE, GM_BF16, nullptr, QB, 256, 1, nullptr, nullptr, 0, 1, 0, 0, 0);
  gemm(LNB, mr_kw, nullptr, 65536, 256, 256, 1.f, ACT_NONE, GM_BF16, nullptr, KB, 256, 1, nullptr, nullptr, 0, 1, 0, 0, 0);
  gemm(LNB, mr_vw, nullptr, 65536, 256, 256, 1.f, ACT_NONE, GM_BF16, nullptr, VB, 256, 1, nullptr, nullptr, 0, 1, 0, 0, 0);
  gemm(LNB, mr_gw, F(8), 65536, 256, 256, 1.f, ACT_SIG, GM_F32, GATEf, nullptr, 256, 1, nullptr, nullptr, 0, 1, 0, 0, 0);
  flash(QB, KB, VB, msa_mask, 0, NBf, 8, GATEf, WGb, 8, 256);
  gemm(WGb, mr_ow, F(10), 65536, 256, 256, 1.f, ACT_NONE, GM_ADDF32, msa, nullptr, 256, 1, nullptr, nullptr, 0, 1, 0, 0, 0);

  // ---- 2. MSA column attention (transposed view) ----
  ln(msa, 23, 24, LNB, 65536, 256, 1);
  gemm(LNB, mc_qw, nullptr, 65536, 256, 256, QS, ACT_NONE, GM_BF16, nullptr, QB, 256, 1, nullptr, nullptr, 0, 1, 0, 0, 0);
  gemm(LNB, mc_kw, nullptr, 65536, 256, 256, 1.f, ACT_NONE, GM_BF16, nullptr, KB, 256, 1, nullptr, nullptr, 0, 1, 0, 0, 0);
  gemm(LNB, mc_vw, nullptr, 65536, 256, 256, 1.f, ACT_NONE, GM_BF16, nullptr, VB, 256, 1, nullptr, nullptr, 0, 1, 0, 0, 0);
  gemm(LNB, mc_gw, F(20), 65536, 256, 256, 1.f, ACT_SIG, GM_F32, GATEf, nullptr, 256, 1, nullptr, nullptr, 0, 1, 0, 0, 0);
  flash(QB, KB, VB, msa_mask, 1, nullptr, 1, GATEf, WGb, 8, 256);
  gemm(WGb, mc_ow, F(22), 65536, 256, 256, 1.f, ACT_NONE, GM_ADD_SWAP, msa, nullptr, 256, 1, nullptr, nullptr, 0, 1, 0, 0, 0);

  // ---- 3. MSA transition ----
  ln(msa, 25, 26, LNB, 65536, 256, 0);
  gemm(LNB, mt_w1, F(28), 65536, 1024, 256, 1.f, ACT_RELU, GM_BF16, nullptr, BIGH, 1024, 1, nullptr, nullptr, 0, 1, 0, 0, 0);
  gemm(BIGH, mt_w2, F(30), 65536, 256, 1024, 1.f, ACT_NONE, GM_ADDF32, msa, nullptr, 256, 1, nullptr, nullptr, 0, 1, 0, 0, 0);

  // ---- 4. Outer product mean -> pair ----
  ln(msa, 31, 32, LNB, 65536, 256, 0);
  gemm(LNB, opm_lw, F(34), 65536, 32, 256, 1.f, ACT_NONE, GM_F32, opmL, nullptr, 32, 1, msa_mask, nullptr, 0, 1, 0, 0, 0);
  gemm(LNB, opm_rw, F(36), 65536, 32, 256, 1.f, ACT_NONE, GM_F32, opmR, nullptr, 32, 1, msa_mask, nullptr, 0, 1, 0, 0, 0);
  k_permopm<<<8192, 256, 0, stream>>>(opmL, opmLT);
  k_permopm<<<8192, 256, 0, stream>>>(opmR, opmRT);
  gemm(opmLT, opmRT, nullptr, 8192, 8192, 256, 1.f, ACT_NONE, GM_OPM, nullptr, BIGH, 0, 0, nullptr, nullptr, 1, 1, 0, 0, 0);
  k_rnorm<<<256, 256, 0, stream>>>(msa_mask, RNf);
  gemm(BIGH, opm_ow, F(38), 65536, 128, 1024, 1.f, ACT_NONE, GM_ADDF32, pair, nullptr, 128, 1, RNf, nullptr, 0, 1, 0, 0, 0);

  // ---- 5. Triangle multiplication, outgoing: out[i,j,c] = sum_k L[i,k,c] R[j,k,c] ----
  ln(pair, 39, 40, LNB, 65536, 128, 0);
  gemm(LNB, tmo_lgw, F(46), 65536, 128, 128, 1.f, ACT_SIG, GM_F32, GATEf, nullptr, 128, 1, nullptr, nullptr, 0, 1, 0, 0, 0);
  gemm(LNB, tmo_lw, F(42), 65536, 128, 128, 1.f, ACT_NONE, GM_F32, triTmp, nullptr, 128, 1, pair_mask, GATEf, 0, 1, 0, 0, 0);
  k_permcm<<<32768, 256, 0, stream>>>(triTmp, triA, 128, 0);
  gemm(LNB, tmo_rgw, F(48), 65536, 128, 128, 1.f, ACT_SIG, GM_F32, GATEf, nullptr, 128, 1, nullptr, nullptr, 0, 1, 0, 0, 0);
  gemm(LNB, tmo_rw, F(44), 65536, 128, 128, 1.f, ACT_NONE, GM_F32, triTmp, nullptr, 128, 1, pair_mask, GATEf, 0, 1, 0, 0, 0);
  k_permcm<<<32768, 256, 0, stream>>>(triTmp, triB, 128, 0);
  gemm(triA, triB, nullptr, 256, 256, 256, 1.f, ACT_NONE, GM_F32, triO, nullptr, 32768, 128, nullptr, nullptr, 1, 128, 65536, 65536, 1);
  gemm(LNB, tmo_gw, F(54), 65536, 128, 128, 1.f, ACT_SIG, GM_F32, GATEf, nullptr, 128, 1, nullptr, nullptr, 0, 1, 0, 0, 0);
  ln(triO, 49, 50, LNB, 65536, 128, 0);
  gemm(LNB, tmo_ow, F(52), 65536, 128, 128, 1.f, ACT_NONE, GM_ADDF32, pair, nullptr, 128, 1, nullptr, GATEf, 0, 1, 0, 0, 0);

  // ---- 6. Triangle multiplication, incoming: out[i,j,c] = sum_k L[k,j,c] R[k,i,c] ----
  ln(pair, 55, 56, LNB, 65536, 128, 0);
  gemm(LNB, tmi_rgw, F(64), 65536, 128, 128, 1.f, ACT_SIG, GM_F32, GATEf, nullptr, 128, 1, nullptr, nullptr, 0, 1, 0, 0, 0);
  gemm(LNB, tmi_rw, F(60), 65536, 128, 128, 1.f, ACT_NONE, GM_F32, triTmp, nullptr, 128, 1, pair_mask, GATEf, 0, 1, 0, 0, 0);
  k_permcm<<<32768, 256, 0, stream>>>(triTmp, triA, 128, 1);   // A_c[i][k] = R[k,i,c]
  gemm(LNB, tmi_lgw, F(62), 65536, 128, 128, 1.f, ACT_SIG, GM_F32, GATEf, nullptr, 128, 1, nullptr, nullptr, 0, 1, 0, 0, 0);
  gemm(LNB, tmi_lw, F(58), 65536, 128, 128, 1.f, ACT_NONE, GM_F32, triTmp, nullptr, 128, 1, pair_mask, GATEf, 0, 1, 0, 0, 0);
  k_permcm<<<32768, 256, 0, stream>>>(triTmp, triB, 128, 1);   // Bt_c[j][k] = L[k,j,c]
  gemm(triA, triB, nullptr, 256, 256, 256, 1.f, ACT_NONE, GM_F32, triO, nullptr, 32768, 128, nullptr, nullptr, 1, 128, 65536, 65536, 1);
  gemm(LNB, tmi_gw, F(70), 65536, 128, 128, 1.f, ACT_SIG, GM_F32, GATEf, nullptr, 128, 1, nullptr, nullptr, 0, 1, 0, 0, 0);
  ln(triO, 65, 66, LNB, 65536, 128, 0);
  gemm(LNB, tmi_ow, F(68), 65536, 128, 128, 1.f, ACT_NONE, GM_ADDF32, pair, nullptr, 128, 1, nullptr, GATEf, 0, 1, 0, 0, 0);

  // ---- 7. Triangle attention, starting node ----
  ln(pair, 78, 79, LNB, 65536, 128, 0);
  gemm(LNB, tas_f2d, nullptr, 65536, 4, 128, 1.f, ACT_NONE, GM_F32, NBf, nullptr, 4, 1, nullptr, nullptr, 0, 1, 0, 0, 0);
  gemm(LNB, tas_qw, nullptr, 65536, 128, 128, QS, ACT_NONE, GM_BF16, nullptr, QB, 128, 1, nullptr, nullptr, 0, 1, 0, 0, 0);
  gemm(LNB, tas_kw, nullptr, 65536, 128, 128, 1.f, ACT_NONE, GM_BF16, nullptr, KB, 128, 1, nullptr, nullptr, 0, 1, 0, 0, 0);
  gemm(LNB, tas_vw, nullptr, 65536, 128, 128, 1.f, ACT_NONE, GM_BF16, nullptr, VB, 128, 1, nullptr, nullptr, 0, 1, 0, 0, 0);
  gemm(LNB, tas_gw, F(75), 65536, 128, 128, 1.f, ACT_SIG, GM_F32, GATEf, nullptr, 128, 1, nullptr, nullptr, 0, 1, 0, 0, 0);
  flash(QB, KB, VB, pair_mask, 0, NBf, 4, GATEf, WGb, 4, 128);
  gemm(WGb, tas_ow, F(77), 65536, 128, 128, 1.f, ACT_NONE, GM_ADDF32, pair, nullptr, 128, 1, nullptr, nullptr, 0, 1, 0, 0, 0);

  // ---- 8. Triangle attention, ending node (transposed view) ----
  ln(pair, 88, 89, LNB, 65536, 128, 1);
  gemm(LNB, tae_f2d, nullptr, 65536, 4, 128, 1.f, ACT_NONE, GM_F32, NBf, nullptr, 4, 1, nullptr, nullptr, 0, 1, 0, 0, 0);
  gemm(LNB, tae_qw, nullptr, 65536, 128, 128, QS, ACT_NONE, GM_BF16, nullptr, QB, 128, 1, nullptr, nullptr, 0, 1, 0, 0, 0);
  gemm(LNB, tae_kw, nullptr, 65536, 128, 128, 1.f, ACT_NONE, GM_BF16, nullptr, KB, 128, 1, nullptr, nullptr, 0, 1, 0, 0, 0);
  gemm(LNB, tae_vw, nullptr, 65536, 128, 128, 1.f, ACT_NONE, GM_BF16, nullptr, VB, 128, 1, nullptr, nullptr, 0, 1, 0, 0, 0);
  gemm(LNB, tae_gw, F(85), 65536, 128, 128, 1.f, ACT_SIG, GM_F32, GATEf, nullptr, 128, 1, nullptr, nullptr, 0, 1, 0, 0, 0);
  flash(QB, KB, VB, pair_mask, 1, NBf, 4, GATEf, WGb, 4, 128);
  gemm(WGb, tae_ow, F(87), 65536, 128, 128, 1.f, ACT_NONE, GM_ADD_SWAP, pair, nullptr, 128, 1, nullptr, nullptr, 0, 1, 0, 0, 0);

  // ---- 9. Pair transition ----
  ln(pair, 91, 92, LNB, 65536, 128, 0);
  gemm(LNB, pt_w1, F(94), 65536, 512, 128, 1.f, ACT_RELU, GM_BF16, nullptr, BIGH, 512, 1, nullptr, nullptr, 0, 1, 0, 0, 0);
  gemm(BIGH, pt_w2, F(96), 65536, 128, 512, 1.f, ACT_NONE, GM_ADDF32, pair, nullptr, 128, 1, nullptr, nullptr, 0, 1, 0, 0, 0);
}
